// StateSpaceMixer_52699248721924
// MI455X (gfx1250) — matrix operand order repaired and checked by reading, not compiled, not run on hardware
//
#include <hip/hip_runtime.h>
#include <hip/hip_bf16.h>

// Problem constants (fixed by the reference)
#define DIMD   1024
#define BATCH  4
#define SEQ    8192
#define MTOT   (BATCH * SEQ)      // 32768 rows for both GEMMs
#define CHUNK  128                // scan chunk length
#define NCHUNK (SEQ / CHUNK)      // 64 chunks

typedef __bf16 bf16;
typedef bf16  v16bf  __attribute__((ext_vector_type(16)));
typedef bf16  bf16x8 __attribute__((ext_vector_type(8)));
typedef bf16  bf16x2 __attribute__((ext_vector_type(2)));
typedef float v8f    __attribute__((ext_vector_type(8)));
typedef unsigned int u32x4 __attribute__((ext_vector_type(4)));
typedef int   i32x4  __attribute__((ext_vector_type(4)));
typedef int   i32x8  __attribute__((ext_vector_type(8)));

// ---------------------------------------------------------------------------
// Prep: convert both weight matrices to bf16, compute A = sigmoid(a)
// ---------------------------------------------------------------------------
__global__ void prep_kernel(const float* __restrict__ w_in,
                            const float* __restrict__ w_out,
                            const float* __restrict__ a,
                            bf16* __restrict__ w_in_bf,
                            bf16* __restrict__ w_out_bf,
                            float* __restrict__ Avec) {
    int i = blockIdx.x * blockDim.x + threadIdx.x;
    if (i < DIMD * DIMD) {
        w_in_bf[i]  = (bf16)w_in[i];
        w_out_bf[i] = (bf16)w_out[i];
    }
    if (i < DIMD) {
        Avec[i] = 1.0f / (1.0f + __expf(-a[i]));
    }
}

// ---------------------------------------------------------------------------
// Staging helpers: 32 contiguous elements per thread -> 4 x bf16x8 registers
// ---------------------------------------------------------------------------
template <typename InT>
__device__ inline void load_stage32(const InT* __restrict__ p, bf16x8 st[4]) {
    if constexpr (sizeof(InT) == 4) {
#pragma unroll
        for (int v = 0; v < 8; ++v) {
            const float4 t = ((const float4*)p)[v];
            st[v >> 1][(v & 1) * 4 + 0] = (bf16)t.x;
            st[v >> 1][(v & 1) * 4 + 1] = (bf16)t.y;
            st[v >> 1][(v & 1) * 4 + 2] = (bf16)t.z;
            st[v >> 1][(v & 1) * 4 + 3] = (bf16)t.w;
        }
    } else {
#pragma unroll
        for (int v = 0; v < 4; ++v) st[v] = ((const bf16x8*)p)[v];
    }
}

__device__ inline void store_stage32(bf16* dst, const bf16x8 st[4]) {
#pragma unroll
    for (int v = 0; v < 4; ++v) ((bf16x8*)dst)[v] = st[v];
}

// ---------------------------------------------------------------------------
// TDM: DMA a 128-row x 64-col bf16 tile (row stride DIMD elems in memory)
// into LDS at lds_byte_off with a 32B pad every 128B row -> LDS row stride
// 160B, matching LDA=80 below. Issued by one wave; tracked with TENSORcnt.
// ---------------------------------------------------------------------------
__device__ inline void tdm_load_tile_128x64(const bf16* gptr, unsigned lds_byte_off) {
    const unsigned long long ga = (unsigned long long)(uintptr_t)gptr;
    u32x4 g0;
    g0[0] = 1u;                                     // count=1 valid descriptor
    g0[1] = lds_byte_off;                           // lds_addr (bytes)
    g0[2] = (unsigned)ga;                           // global_addr[31:0]
    g0[3] = (unsigned)((ga >> 32) & 0x1FFFFFFu)     // global_addr[56:32]
            | 0x80000000u;                          // type=2 ("image")
    i32x8 g1;
    g1[0] = (1 << 16)      // data_size = 2 bytes
          | (1 << 20)      // pad_enable
          | (4 << 22)      // pad_interval: 32 DWORDs (=128B = one tile row)
          | (7 << 25);     // pad_amount: 8 DWORDs (=32B)
    g1[1] = (int)(1024u << 16);          // tensor_dim0[15:0] = 1024 (elems)
    g1[2] = (int)(1024u << 16);          // dim0 hi=0 | tensor_dim1[15:0] = 1024
    g1[3] = (int)(64u << 16);            // dim1 hi=0 | tile_dim0 = 64
    g1[4] = 128;                         // tile_dim1 = 128, tile_dim2 = 0
    g1[5] = 1024;                        // tensor_dim0_stride low32 = 1024
    g1[6] = 0;                           // stride hi | tensor_dim1_stride lo
    g1[7] = 0;
    const i32x4 z4 = {0, 0, 0, 0};
#if defined(__clang_major__) && __clang_major__ >= 23
    const i32x8 z8 = {0, 0, 0, 0, 0, 0, 0, 0};
    __builtin_amdgcn_tensor_load_to_lds(g0, g1, z4, z4, z8, 0);
#else
    __builtin_amdgcn_tensor_load_to_lds(g0, g1, z4, z4, 0);
#endif
}

// ---------------------------------------------------------------------------
// GEMM: C[m,n] = sum_k A[m,k] * W[n,k] + bias[n]
//   A: [MTOT, DIMD] (InT = float or bf16), W: [DIMD, DIMD] bf16 row-major.
// Block tile 128(M) x 128(N), BK=64, double-buffered LDS (1 barrier/step).
// W operand: TDM tensor_load_to_lds DMA (wave 0), overlapped with compute.
// A operand: if f32 -> register-staged convert; if bf16 -> TDM DMA (wave 1).
// 8 waves (2x4); each wave owns 64x32 = 4x2 WMMA tiles -> 16 wmma / K-step.
// Epilogue via LDS for fully coalesced b128 stores.
// ---------------------------------------------------------------------------
template <typename InT, typename OutT>
__global__ __launch_bounds__(256) void gemm_wmma_kernel(
    const InT* __restrict__ Ain, const bf16* __restrict__ W,
    const float* __restrict__ bias, OutT* __restrict__ Cout) {
    constexpr int BM = 128, BN = 128, BK = 64;
    constexpr int LDA = BK + 16;          // 80 elems = 160B row stride (16B-mult)
    constexpr int ASZ = BM * LDA;         // elems per A buffer
    constexpr int BSZ = BN * LDA;         // elems per B buffer
    constexpr int LDC = BN + 4;           // f32 epilogue stride (16B-mult)
    constexpr bool A_VIA_TDM = (sizeof(InT) == 2);

    __shared__ __align__(16) char smem[2 * (ASZ + BSZ) * (int)sizeof(bf16)];
    bf16* sbase = (bf16*)smem;
    const unsigned lds_base = (unsigned)(uintptr_t)sbase;

    const int tid  = threadIdx.x;
    const int lane = tid & 31;
    const int wave = tid >> 5;    // 0..7
    const int wm   = wave >> 2;   // 0..1 -> M offset 64*wm
    const int wn   = wave & 3;    // 0..3 -> N offset 32*wn
    const int m0   = blockIdx.x * BM;
    const int n0   = blockIdx.y * BN;
    const int half = lane >> 4;   // K-half selector per ISA A/B layout
    const int l16  = lane & 15;

    v8f acc[4][2] = {};

    // Cooperative A fill (f32 path): 128 rows x 64 cols, 32 elems per thread
    const int frow = tid >> 1;
    const int fcol = (tid & 1) * 32;
    const InT*  aptr  = Ain + (size_t)(m0 + frow) * DIMD + fcol;
    const bf16* atile = (const bf16*)Ain + (size_t)m0 * DIMD;  // bf16 path
    const bf16* wtile = W + (size_t)n0 * DIMD;                 // + k0 per step

    // Prologue: buffer 0
    if (wave == 0)
        tdm_load_tile_128x64(wtile, lds_base + (unsigned)(ASZ) * 2u);
    bf16x8 ast[4];
    if constexpr (A_VIA_TDM) {
        if (wave == 1)
            tdm_load_tile_128x64(atile, lds_base);
    } else {
        load_stage32<InT>(aptr, ast);
        store_stage32(sbase + frow * LDA + fcol, ast);
    }
    if (wave < 2) __builtin_amdgcn_s_wait_tensorcnt(0);
    __syncthreads();

    constexpr int NK = DIMD / BK;  // 16
#pragma unroll 2
    for (int kb = 0; kb < NK; ++kb) {
        const int cur = kb & 1;
        const bf16* Asb = sbase + cur * (ASZ + BSZ);
        const bf16* Bsb = Asb + ASZ;

        if (kb + 1 < NK) {  // next tiles in flight during compute
            const unsigned nxt_off = (unsigned)((cur ^ 1) * (ASZ + BSZ));
            if (wave == 0)
                tdm_load_tile_128x64(wtile + (kb + 1) * BK,
                                     lds_base + (nxt_off + ASZ) * 2u);
            if constexpr (A_VIA_TDM) {
                if (wave == 1)
                    tdm_load_tile_128x64(atile + (kb + 1) * BK,
                                         lds_base + nxt_off * 2u);
            } else {
                load_stage32<InT>(aptr + (kb + 1) * BK, ast);
            }
        }

#pragma unroll
        for (int ks = 0; ks < BK; ks += 32) {
            v16bf bfrag[2];
#pragma unroll
            for (int j = 0; j < 2; ++j) {
                const bf16* bp = &Bsb[(wn * 32 + j * 16 + l16) * LDA + ks];
                bf16x8 lo = *(const bf16x8*)(bp + half * 8);
                bf16x8 hi = *(const bf16x8*)(bp + 16 + half * 8);
                bfrag[j] = __builtin_shufflevector(lo, hi, 0, 1, 2, 3, 4, 5, 6, 7,
                                                   8, 9, 10, 11, 12, 13, 14, 15);
            }
#pragma unroll
            for (int i = 0; i < 4; ++i) {
                const bf16* ap = &Asb[(wm * 64 + i * 16 + l16) * LDA + ks];
                bf16x8 lo = *(const bf16x8*)(ap + half * 8);
                bf16x8 hi = *(const bf16x8*)(ap + 16 + half * 8);
                v16bf afrag = __builtin_shufflevector(lo, hi, 0, 1, 2, 3, 4, 5, 6, 7,
                                                      8, 9, 10, 11, 12, 13, 14, 15);
#pragma unroll
                for (int j = 0; j < 2; ++j)
                    acc[i][j] = __builtin_amdgcn_wmma_f32_16x16x32_bf16(
                        false, afrag, false, bfrag[j], (short)0, acc[i][j],
                        false, false);
            }
        }

        if (kb + 1 < NK) {
            if constexpr (!A_VIA_TDM)
                store_stage32(sbase + (cur ^ 1) * (ASZ + BSZ) + frow * LDA + fcol, ast);
            if (wave < 2) __builtin_amdgcn_s_wait_tensorcnt(0);
        }
        __syncthreads();
    }

    // ---- Epilogue: accs -> LDS (f32) -> coalesced row-major global stores ----
    float* Cs = (float*)smem;
#pragma unroll
    for (int i = 0; i < 4; ++i) {
#pragma unroll
        for (int j = 0; j < 2; ++j) {
            const int col = wn * 32 + j * 16 + l16;
            const float bv = bias[n0 + col];
#pragma unroll
            for (int r = 0; r < 8; ++r)
                Cs[(wm * 64 + i * 16 + half * 8 + r) * LDC + col] = acc[i][j][r] + bv;
        }
    }
    __syncthreads();

    const int crow = tid >> 1;          // 0..127
    const int cseg = (tid & 1) * 64;    // 64 f32 per thread
    const float* sp = &Cs[crow * LDC + cseg];
    if constexpr (sizeof(OutT) == 4) {
        float4* gp = (float4*)&Cout[(size_t)(m0 + crow) * DIMD + n0 + cseg];
#pragma unroll
        for (int v = 0; v < 16; ++v) gp[v] = ((const float4*)sp)[v];
    } else {
        bf16* gp = (bf16*)&Cout[(size_t)(m0 + crow) * DIMD + n0 + cseg];
#pragma unroll
        for (int v = 0; v < 8; ++v) {
            bf16x8 o;
#pragma unroll
            for (int e = 0; e < 8; ++e) o[e] = (bf16)sp[v * 8 + e];
            ((bf16x8*)gp)[v] = o;
        }
    }
}

// ---------------------------------------------------------------------------
// Parallel linear scan: state_t = A[d]*state_{t-1} + b[d]*h_t, 3-pass chunked.
// Each thread owns TWO adjacent channels (b32 h loads, b32 y stores).
// ---------------------------------------------------------------------------
__global__ __launch_bounds__(256) void scan_partial_kernel(
    const bf16* __restrict__ h, const float* __restrict__ Avec,
    const float* __restrict__ bvec, float* __restrict__ part) {
    int idx = blockIdx.x * blockDim.x + threadIdx.x;  // B*NCHUNK*DIMD/2
    int d = (idx & (DIMD / 2 - 1)) * 2;
    int c = (idx >> 9) & (NCHUNK - 1);
    int b = idx >> 15;
    const float A0 = Avec[d], A1 = Avec[d + 1];
    const float b0 = bvec[d], b1 = bvec[d + 1];
    const bf16* hp = h + ((size_t)b * SEQ + (size_t)c * CHUNK) * DIMD + d;
    float s0 = 0.f, s1 = 0.f;
    for (int i = 0; i < CHUNK; ++i) {
        bf16x2 hv = *(const bf16x2*)&hp[(size_t)i * DIMD];
        s0 = fmaf(A0, s0, b0 * (float)hv[0]);
        s1 = fmaf(A1, s1, b1 * (float)hv[1]);
    }
    float2* pp = (float2*)&part[((size_t)b * NCHUNK + c) * DIMD + d];
    *pp = make_float2(s0, s1);
}

__global__ __launch_bounds__(256) void scan_combine_kernel(
    const float* __restrict__ part, const float* __restrict__ Avec,
    float* __restrict__ carry) {
    int idx = blockIdx.x * blockDim.x + threadIdx.x;  // B*DIMD = 4096
    int d = idx & (DIMD - 1);
    int b = idx >> 10;
    float ap = Avec[d];
#pragma unroll
    for (int i = 0; i < 7; ++i) ap *= ap;  // A^128 via squaring
    float s = 0.f;
    for (int c = 0; c < NCHUNK; ++c) {
        size_t o = ((size_t)b * NCHUNK + c) * DIMD + d;
        carry[o] = s;             // carry-in for chunk c
        s = fmaf(ap, s, part[o]);
    }
}

__global__ __launch_bounds__(256) void scan_final_kernel(
    const bf16* __restrict__ h, const float* __restrict__ Avec,
    const float* __restrict__ bvec, const float* __restrict__ carry,
    bf16* __restrict__ y) {
    int idx = blockIdx.x * blockDim.x + threadIdx.x;  // B*NCHUNK*DIMD/2
    int d = (idx & (DIMD / 2 - 1)) * 2;
    int c = (idx >> 9) & (NCHUNK - 1);
    int b = idx >> 15;
    const float A0 = Avec[d], A1 = Avec[d + 1];
    const float b0 = bvec[d], b1 = bvec[d + 1];
    const size_t base = ((size_t)b * SEQ + (size_t)c * CHUNK) * DIMD + d;
    const bf16* hp = h + base;
    bf16* yp = y + base;
    const float2 cv = *(const float2*)&carry[((size_t)b * NCHUNK + c) * DIMD + d];
    float s0 = cv.x, s1 = cv.y;
    for (int i = 0; i < CHUNK; ++i) {
        bf16x2 hv = *(const bf16x2*)&hp[(size_t)i * DIMD];
        s0 = fmaf(A0, s0, b0 * (float)hv[0]);
        s1 = fmaf(A1, s1, b1 * (float)hv[1]);
        bf16x2 ov;
        ov[0] = (bf16)s0;
        ov[1] = (bf16)s1;
        *(bf16x2*)&yp[(size_t)i * DIMD] = ov;
    }
}

// ---------------------------------------------------------------------------
extern "C" void kernel_launch(void* const* d_in, const int* in_sizes, int n_in,
                              void* d_out, int out_size, void* d_ws,
                              size_t ws_size, hipStream_t stream) {
    const float* x     = (const float*)d_in[0];  // [B,T,D]
    const float* w_in  = (const float*)d_in[1];  // [D,D]
    const float* b_in  = (const float*)d_in[2];  // [D]
    const float* a     = (const float*)d_in[3];  // [D]
    const float* bvec  = (const float*)d_in[4];  // [D]
    const float* w_out = (const float*)d_in[5];  // [D,D]
    const float* b_out = (const float*)d_in[6];  // [D]
    float* out = (float*)d_out;

    char* ws = (char*)d_ws;
    size_t off = 0;
    auto walloc = [&](size_t bytes) {
        void* p = ws + off;
        off += (bytes + 255) & ~(size_t)255;
        return p;
    };
    bf16*  w_in_bf  = (bf16*)walloc((size_t)DIMD * DIMD * sizeof(bf16));
    bf16*  w_out_bf = (bf16*)walloc((size_t)DIMD * DIMD * sizeof(bf16));
    float* Avec     = (float*)walloc(DIMD * sizeof(float));
    bf16*  hbuf     = (bf16*)walloc((size_t)MTOT * DIMD * sizeof(bf16));
    bf16*  ybuf     = (bf16*)walloc((size_t)MTOT * DIMD * sizeof(bf16));
    float* part     = (float*)walloc((size_t)BATCH * NCHUNK * DIMD * sizeof(float));
    float* carry    = (float*)walloc((size_t)BATCH * NCHUNK * DIMD * sizeof(float));

    prep_kernel<<<(DIMD * DIMD + 255) / 256, 256, 0, stream>>>(
        w_in, w_out, a, w_in_bf, w_out_bf, Avec);

    dim3 gg(MTOT / 128, DIMD / 128);
    gemm_wmma_kernel<float, bf16><<<gg, 256, 0, stream>>>(x, w_in_bf, b_in, hbuf);

    scan_partial_kernel<<<(BATCH * NCHUNK * DIMD / 2) / 256, 256, 0, stream>>>(
        hbuf, Avec, bvec, part);
    scan_combine_kernel<<<(BATCH * DIMD) / 256, 256, 0, stream>>>(part, Avec, carry);
    scan_final_kernel<<<(BATCH * NCHUNK * DIMD / 2) / 256, 256, 0, stream>>>(
        hbuf, Avec, bvec, carry, ybuf);

    gemm_wmma_kernel<bf16, float><<<gg, 256, 0, stream>>>(ybuf, w_out_bf, b_out, out);
}
